// RWKV_Tmix_x070_70334384439672
// MI455X (gfx1250) — compile-verified, matrix operand order checked by reading, not run on hardware
//
#include <hip/hip_runtime.h>
#include <hip/hip_bf16.h>

// ---------------- problem constants (fixed by the reference) ----------------
constexpr int Bq  = 4;
constexpr int Tq  = 2048;
constexpr int Cq  = 2048;
constexpr int Hq  = 32;      // heads
constexpr int HSq = 64;      // head size
constexpr int NBT = Bq * Tq; // 8192 rows
constexpr int LWc = 64, LAc = 64, LVc = 32, LGc = 128;
constexpr float EPS_GN = 0.00064f;

typedef __attribute__((ext_vector_type(16))) __bf16 v16bf;
typedef __attribute__((ext_vector_type(8)))  float  v8f;

// gfx1250 async global->LDS path (guarded: falls back to plain LDS stores)
#if defined(__has_builtin)
# if __has_builtin(__builtin_amdgcn_global_load_async_to_lds_b32) && \
     __has_builtin(__builtin_amdgcn_s_wait_asynccnt)
#  define USE_ASYNC_LDS 1
# endif
#endif

#ifdef USE_ASYNC_LDS
// builtin signature (from hipcc diagnostics): (global int*, shared int*, imm offset, imm cpol)
#define ASYNC_G2L(srcp, dstp)                                              \
    __builtin_amdgcn_global_load_async_to_lds_b32(                         \
        (__attribute__((address_space(1))) int*)(srcp),                    \
        (__attribute__((address_space(3))) int*)(dstp), 0, 0)
#endif

// ---------------- helpers ----------------
__device__ __forceinline__ float bf2f(unsigned short h) {
    return __uint_as_float(((unsigned int)h) << 16);
}
__device__ __forceinline__ unsigned short f2bf(float f) {
    unsigned int u = __float_as_uint(f);
    u += 0x7fffu + ((u >> 16) & 1u);   // round-to-nearest-even
    return (unsigned short)(u >> 16);
}
__device__ __forceinline__ float sigmoidf_(float x) {
    return 1.f / (1.f + __expf(-x));
}
__device__ __forceinline__ float softplusf_(float t) {
    return (t > 20.f) ? t : log1pf(__expf(t));
}

// ---------------- 1) transpose+convert weight: W[K,N] f32 -> WT[N,K] bf16 ---
__global__ void cvtT_kernel(const float* __restrict__ W,
                            unsigned short* __restrict__ WT, int K, int N) {
    size_t e = (size_t)blockIdx.x * blockDim.x + threadIdx.x;
    if (e >= (size_t)K * N) return;
    int k = (int)(e / N), n = (int)(e % N);
    WT[(size_t)n * K + k] = f2bf(W[e]);
}

// ---------------- 2) token shift -> six bf16 mixed activations --------------
__global__ void shift_kernel(const float* __restrict__ x,
                             const float* __restrict__ x_prev,
                             const float* __restrict__ mr, const float* __restrict__ mw,
                             const float* __restrict__ mk, const float* __restrict__ mv,
                             const float* __restrict__ ma, const float* __restrict__ mg,
                             unsigned short* __restrict__ xr, unsigned short* __restrict__ xw,
                             unsigned short* __restrict__ xk, unsigned short* __restrict__ xv,
                             unsigned short* __restrict__ xa, unsigned short* __restrict__ xg) {
    size_t e = (size_t)blockIdx.x * blockDim.x + threadIdx.x;
    if (e >= (size_t)NBT * Cq) return;
    int c = (int)(e % Cq);
    size_t rowt = e / Cq;
    int t = (int)(rowt % Tq);
    int b = (int)(rowt / Tq);
    float xc = x[e];
    float prev = (t == 0) ? x_prev[(size_t)b * Cq + c] : x[e - Cq];
    float xx = prev - xc;
    xr[e] = f2bf(xc + xx * mr[c]);
    xw[e] = f2bf(xc + xx * mw[c]);
    xk[e] = f2bf(xc + xx * mk[c]);
    xv[e] = f2bf(xc + xx * mv[c]);
    xa[e] = f2bf(xc + xx * ma[c]);
    xg[e] = f2bf(xc + xx * mg[c]);
}

// ---------------- 3) bf16 WMMA GEMM: C[M,N] = A[M,K] @ B[K,N] ---------------
// A row-major bf16 (ushort bits), BT = B^T row-major [N,K] bf16, C f32 row-major.
// One wave computes a 64x64 tile via 4x4 v_wmma_f32_16x16x32_bf16 accumulators:
// per K=32 step, 8 fragment loads feed 16 WMMAs (32 FLOP/byte from cache).
__global__ __launch_bounds__(256) void gemm_bf16_wmma(
        const unsigned short* __restrict__ A,
        const unsigned short* __restrict__ BT,
        float* __restrict__ Cmat,
        int M, int N, int K) {
    const int lane = threadIdx.x & 31;
    const int wave = blockIdx.x * (blockDim.x >> 5) + (threadIdx.x >> 5);
    const int tilesN = N >> 6;
    const int tm = wave / tilesN;
    const int tn = wave % tilesN;
    if (tm * 64 >= M) return;

    const int lr = lane & 15;          // row/col within a 16-tile
    const int kh = (lane >> 4) << 4;   // K half offset: 0 or 16
    size_t rowA[4], colB[4];
#pragma unroll
    for (int i = 0; i < 4; ++i) {
        rowA[i] = (size_t)(tm * 64 + i * 16 + lr) * K;
        colB[i] = (size_t)(tn * 64 + i * 16 + lr) * K;
    }

    v8f acc[4][4] = {};

    for (int kb = 0; kb < K; kb += 32) {
        const int ko = kb + kh;
        v16bf av[4], bv[4];
#pragma unroll
        for (int i = 0; i < 4; ++i) av[i] = *(const v16bf*)(A  + rowA[i] + ko);
#pragma unroll
        for (int i = 0; i < 4; ++i) bv[i] = *(const v16bf*)(BT + colB[i] + ko);
#pragma unroll
        for (int mi = 0; mi < 4; ++mi)
#pragma unroll
            for (int ni = 0; ni < 4; ++ni)
                acc[mi][ni] = __builtin_amdgcn_wmma_f32_16x16x32_bf16(
                    false, av[mi], false, bv[ni], (short)0, acc[mi][ni], false, false);
    }

    // C/D layout: lanes 0-15 -> N=lane, M=v; lanes 16-31 -> N=lane-16, M=8+v
    const int colOff = lane & 15;
    const int rowOff = (lane >> 4) << 3;
#pragma unroll
    for (int mi = 0; mi < 4; ++mi) {
#pragma unroll
        for (int ni = 0; ni < 4; ++ni) {
            const int r0 = tm * 64 + mi * 16 + rowOff;
            const int c0 = tn * 64 + ni * 16 + colOff;
#pragma unroll
            for (int v = 0; v < 8; ++v)
                Cmat[(size_t)(r0 + v) * N + c0] = acc[mi][ni][v];
        }
    }
}

// ---------------- 4) LoRA stage-1 matvecs: out[row,j] = act(xin[row,:] @ W[:,j])
__global__ void mlp1_kernel(const unsigned short* __restrict__ xin,
                            const float* __restrict__ W,
                            float* __restrict__ out, int L, int act) {
    int tid = blockIdx.x * blockDim.x + threadIdx.x;
    int row = tid / L, j = tid % L;
    if (row >= NBT) return;
    const unsigned short* ip = xin + (size_t)row * Cq;
    float acc = 0.f;
    for (int c = 0; c < Cq; ++c) acc += bf2f(ip[c]) * W[c * L + j];
    if (act == 1)      acc = tanhf(acc);
    else if (act == 2) acc = sigmoidf_(acc);
    out[(size_t)row * L + j] = acc;
}

// ---------------- 5) prep: decay, a, v-mix, g, k-adjust, kk normalize -------
// One wave per (row, head); lane covers channels {lane, lane+32} of the head.
__global__ void prep_kernel(float* __restrict__ kbuf, float* __restrict__ vbuf,
                            const float* __restrict__ vfirst,
                            const float* __restrict__ hw, const float* __restrict__ ha,
                            const float* __restrict__ hv, const float* __restrict__ hg,
                            const float* __restrict__ w0, const float* __restrict__ w2,
                            const float* __restrict__ a0, const float* __restrict__ a2,
                            const float* __restrict__ v0, const float* __restrict__ v2,
                            const float* __restrict__ g2,
                            const float* __restrict__ k_k, const float* __restrict__ k_a,
                            float* __restrict__ wdbuf, float* __restrict__ ahbuf,
                            float* __restrict__ bhbuf, float* __restrict__ gbuf) {
    const int wave = threadIdx.x >> 5, lane = threadIdx.x & 31;
    const int gh = blockIdx.x * 8 + wave;        // row*H + h
    const int row = gh >> 5;
    const int h = gh & 31;
    const size_t base = (size_t)row * Cq + (size_t)h * HSq;

    float kk2sum = 0.f;
    float kkv[2], av[2], knew[2], wdv[2], gvv[2];
    const int chs[2] = {lane, lane + 32};

#pragma unroll
    for (int p = 0; p < 2; ++p) {
        const int c = chs[p];
        const int ch = h * HSq + c;
        const size_t idx = base + c;
        const float kval = kbuf[idx];

        float accA = a0[ch];
        for (int j = 0; j < LAc; ++j) accA += ha[row * LAc + j] * a2[j * Cq + ch];
        const float a = sigmoidf_(accA);

        float accW = w0[ch];
        for (int j = 0; j < LWc; ++j) accW += hw[row * LWc + j] * w2[j * Cq + ch];
        const float wraw = -softplusf_(-accW) - 0.5f;
        wdv[p] = __expf(-__expf(wraw));

        float accV = v0[ch];
        for (int j = 0; j < LVc; ++j) accV += hv[row * LVc + j] * v2[j * Cq + ch];
        const float gate = sigmoidf_(accV);
        const float vval = vbuf[idx];
        vbuf[idx] = vval + (vfirst[idx] - vval) * gate;

        float accG = 0.f;
        for (int j = 0; j < LGc; ++j) accG += hg[row * LGc + j] * g2[j * Cq + ch];
        gvv[p] = accG;

        const float kk = kval * k_k[ch];
        kkv[p] = kk;
        av[p] = a;
        kk2sum += kk * kk;
        knew[p] = kval * (1.f + (a - 1.f) * k_a[ch]);
    }
    // wave32 reduction over the head's 64 channels
    for (int m = 16; m >= 1; m >>= 1) kk2sum += __shfl_xor(kk2sum, m, 32);
    const float scale = 1.f / fmaxf(sqrtf(kk2sum), 1e-12f);

#pragma unroll
    for (int p = 0; p < 2; ++p) {
        const size_t idx = base + chs[p];
        const float kkn = kkv[p] * scale;
        kbuf[idx]  = knew[p];
        wdbuf[idx] = wdv[p];
        ahbuf[idx] = -kkn;
        bhbuf[idx] = kkn * av[p];
        gbuf[idx]  = gvv[p];
    }
}

// ---------------- 6) WKV sequential scan: 1 block per (b,h), 64 lanes -------
__global__ __launch_bounds__(64) void wkv_scan_kernel(
        const float* __restrict__ rbuf, const float* __restrict__ wdbuf,
        const float* __restrict__ kbuf, const float* __restrict__ vbuf,
        const float* __restrict__ ahbuf, const float* __restrict__ bhbuf,
        const float* __restrict__ state0, float* __restrict__ outbuf) {
    const int b = blockIdx.x >> 5;
    const int h = blockIdx.x & 31;
    const int i = threadIdx.x;          // state row 0..63

    __shared__ float shr[HSq], shw[HSq], shk[HSq], shv[HSq], sha[HSq], shb[HSq];

    float st[HSq];
    const size_t sbase = (((size_t)(b * Hq + h)) * HSq + i) * HSq;
#pragma unroll
    for (int j = 0; j < HSq; ++j) st[j] = state0[sbase + j];

    size_t base = (size_t)b * Tq * Cq + (size_t)h * HSq;
    for (int t = 0; t < Tq; ++t) {
#ifdef USE_ASYNC_LDS
        // CDNA5 async global->LDS staging (ASYNCcnt-tracked)
        ASYNC_G2L(rbuf  + base + i, &shr[i]);
        ASYNC_G2L(wdbuf + base + i, &shw[i]);
        ASYNC_G2L(kbuf  + base + i, &shk[i]);
        ASYNC_G2L(vbuf  + base + i, &shv[i]);
        ASYNC_G2L(ahbuf + base + i, &sha[i]);
        ASYNC_G2L(bhbuf + base + i, &shb[i]);
        __builtin_amdgcn_s_wait_asynccnt(0);
#else
        shr[i] = rbuf[base + i];
        shw[i] = wdbuf[base + i];
        shk[i] = kbuf[base + i];
        shv[i] = vbuf[base + i];
        sha[i] = ahbuf[base + i];
        shb[i] = bhbuf[base + i];
#endif
        __syncthreads();

        if (t + 1 < Tq) {   // global_prefetch_b8 for next timestep
            __builtin_prefetch(rbuf  + base + Cq + i, 0, 1);
            __builtin_prefetch(wdbuf + base + Cq + i, 0, 1);
            __builtin_prefetch(kbuf  + base + Cq + i, 0, 1);
            __builtin_prefetch(ahbuf + base + Cq + i, 0, 1);
        }

        float sa = 0.f;
#pragma unroll
        for (int j = 0; j < HSq; ++j) sa += st[j] * sha[j];

        const float vi = shv[i];
        float o = 0.f;
#pragma unroll
        for (int j = 0; j < HSq; ++j) {
            const float s = st[j] * shw[j] + sa * shb[j] + vi * shk[j];
            st[j] = s;
            o += s * shr[j];
        }
        outbuf[base + i] = o;
        __syncthreads();
        base += Cq;
    }
}

// ---------------- 7) GroupNorm + bonus + gate -> bf16 -----------------------
__global__ void post_kernel(const float* __restrict__ outbuf,
                            const float* __restrict__ rbuf, const float* __restrict__ kbuf,
                            const float* __restrict__ vbuf, const float* __restrict__ gbuf,
                            const float* __restrict__ r_k,
                            const float* __restrict__ ln_w, const float* __restrict__ ln_b,
                            unsigned short* __restrict__ ybf) {
    const int wave = threadIdx.x >> 5, lane = threadIdx.x & 31;
    const int gh = blockIdx.x * 8 + wave;
    const int row = gh >> 5;
    const int h = gh & 31;
    const size_t base = (size_t)row * Cq + (size_t)h * HSq;

    const int chs[2] = {lane, lane + 32};
    float o[2];
    float sum = 0.f, sumsq = 0.f, dot = 0.f;
#pragma unroll
    for (int p = 0; p < 2; ++p) {
        const size_t idx = base + chs[p];
        o[p] = outbuf[idx];
        sum += o[p];
        sumsq += o[p] * o[p];
        dot += rbuf[idx] * kbuf[idx] * r_k[h * HSq + chs[p]];
    }
    for (int m = 16; m >= 1; m >>= 1) {
        sum   += __shfl_xor(sum, m, 32);
        sumsq += __shfl_xor(sumsq, m, 32);
        dot   += __shfl_xor(dot, m, 32);
    }
    const float mu  = sum * (1.f / 64.f);
    const float var = sumsq * (1.f / 64.f) - mu * mu;
    const float inv = rsqrtf(var + EPS_GN);
#pragma unroll
    for (int p = 0; p < 2; ++p) {
        const int ch = h * HSq + chs[p];
        const size_t idx = base + chs[p];
        float y = (o[p] - mu) * inv * ln_w[ch] + ln_b[ch];
        y += dot * vbuf[idx];
        ybf[idx] = f2bf(y * gbuf[idx]);
    }
}

// ---------------- launcher ----------------
extern "C" void kernel_launch(void* const* d_in, const int* in_sizes, int n_in,
                              void* d_out, int out_size, void* d_ws, size_t ws_size,
                              hipStream_t stream) {
    (void)in_sizes; (void)n_in; (void)out_size; (void)ws_size;

    const float* x       = (const float*)d_in[0];
    const float* v_first = (const float*)d_in[1];
    const float* x_prev  = (const float*)d_in[2];
    const float* wkv_st  = (const float*)d_in[3];
    const float* m_r = (const float*)d_in[4];
    const float* m_w = (const float*)d_in[5];
    const float* m_k = (const float*)d_in[6];
    const float* m_v = (const float*)d_in[7];
    const float* m_a = (const float*)d_in[8];
    const float* m_g = (const float*)d_in[9];
    const float* w0 = (const float*)d_in[10];
    const float* w1 = (const float*)d_in[11];
    const float* w2 = (const float*)d_in[12];
    const float* a0 = (const float*)d_in[13];
    const float* a1 = (const float*)d_in[14];
    const float* a2 = (const float*)d_in[15];
    const float* v0 = (const float*)d_in[16];
    const float* v1 = (const float*)d_in[17];
    const float* v2 = (const float*)d_in[18];
    const float* g1 = (const float*)d_in[19];
    const float* g2 = (const float*)d_in[20];
    const float* k_k = (const float*)d_in[21];
    const float* k_a = (const float*)d_in[22];
    const float* r_k = (const float*)d_in[23];
    const float* Wr = (const float*)d_in[24];
    const float* Wk = (const float*)d_in[25];
    const float* Wv = (const float*)d_in[26];
    const float* Wo = (const float*)d_in[27];
    const float* ln_w = (const float*)d_in[28];
    const float* ln_b = (const float*)d_in[29];

    float* out = (float*)d_out;

    // ---- workspace layout ----
    const size_t BTC = (size_t)NBT * Cq;          // 16,777,216
    const size_t CC  = (size_t)Cq * Cq;           //  4,194,304
    char* p = (char*)d_ws;
    auto take = [&](size_t bytes) { char* r = p; p += (bytes + 255) & ~(size_t)255; return r; };

    unsigned short* xr = (unsigned short*)take(BTC * 2);
    unsigned short* xw = (unsigned short*)take(BTC * 2);
    unsigned short* xk = (unsigned short*)take(BTC * 2);
    unsigned short* xv = (unsigned short*)take(BTC * 2);
    unsigned short* xa = (unsigned short*)take(BTC * 2);
    unsigned short* xg = (unsigned short*)take(BTC * 2);
    unsigned short* WrT = (unsigned short*)take(CC * 2);
    unsigned short* WkT = (unsigned short*)take(CC * 2);
    unsigned short* WvT = (unsigned short*)take(CC * 2);
    unsigned short* WoT = (unsigned short*)take(CC * 2);
    float* rbuf  = (float*)take(BTC * 4);
    float* kbuf  = (float*)take(BTC * 4);
    float* vbuf  = (float*)take(BTC * 4);
    float* hw    = (float*)take((size_t)NBT * LWc * 4);
    float* ha    = (float*)take((size_t)NBT * LAc * 4);
    float* hv    = (float*)take((size_t)NBT * LVc * 4);
    float* hg    = (float*)take((size_t)NBT * LGc * 4);
    float* wdbuf = (float*)take(BTC * 4);
    float* ahbuf = (float*)take(BTC * 4);
    float* bhbuf = (float*)take(BTC * 4);
    float* gbuf  = (float*)take(BTC * 4);
    float* outbf = (float*)take(BTC * 4);
    unsigned short* ybf = (unsigned short*)take(BTC * 2);

    const int THREADS = 256;

    // 1) weight transpose + bf16 convert
    {
        dim3 g((unsigned)(CC / THREADS));
        cvtT_kernel<<<g, THREADS, 0, stream>>>(Wr, WrT, Cq, Cq);
        cvtT_kernel<<<g, THREADS, 0, stream>>>(Wk, WkT, Cq, Cq);
        cvtT_kernel<<<g, THREADS, 0, stream>>>(Wv, WvT, Cq, Cq);
        cvtT_kernel<<<g, THREADS, 0, stream>>>(Wo, WoT, Cq, Cq);
    }
    // 2) token shift + mixes
    shift_kernel<<<dim3((unsigned)(BTC / THREADS)), THREADS, 0, stream>>>(
        x, x_prev, m_r, m_w, m_k, m_v, m_a, m_g, xr, xw, xk, xv, xa, xg);

    // 3) big bf16 WMMA GEMMs: r, k, v  (64x64 tile per wave)
    {
        const int waves = (NBT / 64) * (Cq / 64);   // 4096
        dim3 g((unsigned)(waves / 8));
        gemm_bf16_wmma<<<g, THREADS, 0, stream>>>(xr, WrT, rbuf, NBT, Cq, Cq);
        gemm_bf16_wmma<<<g, THREADS, 0, stream>>>(xk, WkT, kbuf, NBT, Cq, Cq);
        gemm_bf16_wmma<<<g, THREADS, 0, stream>>>(xv, WvT, vbuf, NBT, Cq, Cq);
    }
    // 4) LoRA stage-1
    mlp1_kernel<<<dim3((unsigned)((size_t)NBT * LWc / THREADS)), THREADS, 0, stream>>>(xw, w1, hw, LWc, 1);
    mlp1_kernel<<<dim3((unsigned)((size_t)NBT * LAc / THREADS)), THREADS, 0, stream>>>(xa, a1, ha, LAc, 0);
    mlp1_kernel<<<dim3((unsigned)((size_t)NBT * LVc / THREADS)), THREADS, 0, stream>>>(xv, v1, hv, LVc, 0);
    mlp1_kernel<<<dim3((unsigned)((size_t)NBT * LGc / THREADS)), THREADS, 0, stream>>>(xg, g1, hg, LGc, 2);

    // 5) prep (decay / a / v-mix / g / kk-norm / k-adjust)
    prep_kernel<<<dim3((unsigned)((size_t)NBT * Hq / 8)), THREADS, 0, stream>>>(
        kbuf, vbuf, v_first, hw, ha, hv, hg, w0, w2, a0, a2, v0, v2, g2,
        k_k, k_a, wdbuf, ahbuf, bhbuf, gbuf);

    // 6) sequential WKV scan: one block per (b,h)
    wkv_scan_kernel<<<dim3(Bq * Hq), 64, 0, stream>>>(
        rbuf, wdbuf, kbuf, vbuf, ahbuf, bhbuf, wkv_st, outbf);

    // 7) GroupNorm + bonus + gate -> bf16
    post_kernel<<<dim3((unsigned)((size_t)NBT * Hq / 8)), THREADS, 0, stream>>>(
        outbf, rbuf, kbuf, vbuf, gbuf, r_k, ln_w, ln_b, ybf);

    // 8) final projection: d_out = y @ Wo
    {
        const int waves = (NBT / 64) * (Cq / 64);
        dim3 g((unsigned)(waves / 8));
        gemm_bf16_wmma<<<g, THREADS, 0, stream>>>(ybf, WoT, out, NBT, Cq, Cq);
    }
}
